// BCPQuantizedLinear_64330020159889
// MI455X (gfx1250) — compile-verified, matrix operand order checked by reading, not run on hardware
//
#include <hip/hip_runtime.h>
#include <stdint.h>

typedef __attribute__((ext_vector_type(16))) _Float16 v16h;
typedef __attribute__((ext_vector_type(8)))  float    v8f;

#define MTOK 64
#define NDIM 8192
#define KDIM 8192
#define GSZ  64
#define NGRP (NDIM / GSZ)                   // 128
#define NSPLIT 4
#define CHUNKS (NDIM / 32)                  // 256 chunks of 32 along N
#define CHUNKS_PER_SPLIT (CHUNKS / NSPLIT)  // 64
#define KPERBLOCK 64                        // 4 waves x 16 features
#define NBUF 4                              // quad-buffered LDS, distance-2 prefetch

// ---------------------------------------------------------------------------
// Async global -> LDS copy (CDNA5, tracked by ASYNCcnt). lds_off is the LDS
// byte offset (low 32 bits of the generic shared-aperture address).
// ---------------------------------------------------------------------------
__device__ __forceinline__ void async_ld_b128(uint32_t lds_off, const void* g) {
    asm volatile("global_load_async_to_lds_b128 %0, %1, off"
                 :: "v"(lds_off), "v"(g) : "memory");
}

// ---------------------------------------------------------------------------
// Prep: x[f32 64x8192] * mu1[n]  ->  f16 A-fragments in WMMA 16x32 lane layout
// A-layout (ISA 7.12.2, 16-bit A 16x32): VGPR j holds K = (j>=4?16:0) +
//   (lane>=16?8:0) + 2*(j&3) + p  for row M = lane%16.
// ---------------------------------------------------------------------------
__global__ void prep_x_kernel(const float* __restrict__ x,
                              const float* __restrict__ mu1,
                              _Float16* __restrict__ xh) {
    int tid = blockIdx.x * blockDim.x + threadIdx.x;   // [0, 262144)
    int j = tid & 7;
    int L = (tid >> 3) & 31;
    int t = (tid >> 8) & 3;
    int c = tid >> 10;
    int m  = t * 16 + (L & 15);
    int nb = c * 32 + ((j & 4) << 2) + ((L >> 4) << 3) + ((j & 3) << 1);
    float2 xv = *(const float2*)(x   + (size_t)m * NDIM + nb);
    float2 mv = *(const float2*)(mu1 + nb);
    xh[(size_t)tid * 2]     = (_Float16)(xv.x * mv.x);
    xh[(size_t)tid * 2 + 1] = (_Float16)(xv.y * mv.y);
}

// ---------------------------------------------------------------------------
// Seed output with bias (N-split partial sums are atomically accumulated).
// ---------------------------------------------------------------------------
__global__ void init_out_kernel(const float* __restrict__ bias,
                                float* __restrict__ out) {
    int tid = blockIdx.x * blockDim.x + threadIdx.x;   // [0, 524288)
    out[tid] = bias[tid & (KDIM - 1)];
}

// ---------------------------------------------------------------------------
// Main: async-stream W_q + mask into quad-buffered LDS (2 chunks in flight),
// dequant to f16, WMMA against L2-resident pre-swizzled x.
// grid = (KDIM/KPERBLOCK, NSPLIT), block = 128 (4 waves).
// ---------------------------------------------------------------------------
__global__ void __launch_bounds__(128)
qlinear_wmma_kernel(const int8_t* __restrict__ Wq,
                    const float*  __restrict__ scales,
                    const float*  __restrict__ zeros,
                    const float*  __restrict__ mask,
                    const float*  __restrict__ mu2,
                    const _Float16* __restrict__ xh,
                    float* __restrict__ out) {
    // bank-conflict-padded, 16B-aligned staging tiles (async DMA targets)
    __shared__ uint8_t s_wq[NBUF][KPERBLOCK][48];   // 32 bytes used per row
    __shared__ float   s_mk[NBUF][KPERBLOCK][36];   // 32 floats used per row

    const int tid    = threadIdx.x;
    const int wave   = tid >> 5;
    const int lane   = tid & 31;
    const int kblock = blockIdx.x * KPERBLOCK;
    const int c0     = blockIdx.y * CHUNKS_PER_SPLIT;   // multiple of 64
    const int cend   = c0 + CHUNKS_PER_SPLIT;

    const int fr = wave * 16 + (lane & 15);     // feature row within block tile
    const int f  = kblock + fr;                 // global output feature
    const int h8 = (lane >> 4) << 3;            // 0 or 8 (lane-half K offset)

    const int strow  = tid >> 1;                // staging: row 0..63
    const int sthalf = tid & 1;                 // staging: which 16-elem half
    const size_t grow = (size_t)(kblock + strow) * NDIM + sthalf * 16;

    // issue one chunk's staging batch: 1 b128 of W_q + 4 b128 of mask / thread
    auto issue = [&](int cc) {
        const int b = cc & (NBUF - 1);
        const size_t gb = grow + (size_t)cc * 32;
        async_ld_b128((uint32_t)(uintptr_t)&s_wq[b][strow][sthalf * 16],      Wq   + gb);
        async_ld_b128((uint32_t)(uintptr_t)&s_mk[b][strow][sthalf * 16 + 0],  mask + gb + 0);
        async_ld_b128((uint32_t)(uintptr_t)&s_mk[b][strow][sthalf * 16 + 4],  mask + gb + 4);
        async_ld_b128((uint32_t)(uintptr_t)&s_mk[b][strow][sthalf * 16 + 8],  mask + gb + 8);
        async_ld_b128((uint32_t)(uintptr_t)&s_mk[b][strow][sthalf * 16 + 12], mask + gb + 12);
    };

    v8f acc0 = {}, acc1 = {}, acc2 = {}, acc3 = {};

    // dequant + 4x WMMA for chunk c (its tile already resident in LDS buf c&3)
    auto compute = [&](int c) {
        const int bi = c & (NBUF - 1);
        const float s  = scales[(size_t)f * NGRP + (c >> 1)];   // group = n0/64
        const float zb = zeros [(size_t)f * NGRP + (c >> 1)] * s;

        uint8_t qa[8], qb[8];
        *(uint2*)qa = *(const uint2*)&s_wq[bi][fr][h8];
        *(uint2*)qb = *(const uint2*)&s_wq[bi][fr][16 + h8];
        float ma[8], mb[8];
        *(float4*)(ma)     = *(const float4*)&s_mk[bi][fr][h8];
        *(float4*)(ma + 4) = *(const float4*)&s_mk[bi][fr][h8 + 4];
        *(float4*)(mb)     = *(const float4*)&s_mk[bi][fr][16 + h8];
        *(float4*)(mb + 4) = *(const float4*)&s_mk[bi][fr][16 + h8 + 4];

        v16h bf;
        #pragma unroll
        for (int e = 0; e < 8; ++e)
            bf[e] = (_Float16)(fmaf((float)qa[e], s, -zb) * ma[e]);
        #pragma unroll
        for (int e = 0; e < 8; ++e)
            bf[8 + e] = (_Float16)(fmaf((float)qb[e], s, -zb) * mb[e]);

        const _Float16* ap = xh + (((size_t)c * 4 * 32) + lane) * 16;
        v16h a0 = *(const v16h*)(ap);
        v16h a1 = *(const v16h*)(ap + 32 * 16);
        v16h a2 = *(const v16h*)(ap + 64 * 16);
        v16h a3 = *(const v16h*)(ap + 96 * 16);

        acc0 = __builtin_amdgcn_wmma_f32_16x16x32_f16(false, a0, false, bf,
                                                      (short)0, acc0, false, false);
        acc1 = __builtin_amdgcn_wmma_f32_16x16x32_f16(false, a1, false, bf,
                                                      (short)0, acc1, false, false);
        acc2 = __builtin_amdgcn_wmma_f32_16x16x32_f16(false, a2, false, bf,
                                                      (short)0, acc2, false, false);
        acc3 = __builtin_amdgcn_wmma_f32_16x16x32_f16(false, a3, false, bf,
                                                      (short)0, acc3, false, false);
    };

    // ---- software pipeline: distance-2 async prefetch, branch-free hot loop --
    issue(c0);
    issue(c0 + 1);
    for (int c = c0; c < cend - 2; ++c) {
        issue(c + 2);                               // keep 2 batches in flight
        asm volatile("s_wait_asynccnt 0xa" ::: "memory");   // drain batch c
        __syncthreads();
        compute(c);
        // one barrier/iter is safe: iter c writes buf (c+2)&3; waves (skew<=1)
        // read only bufs (c-1)&3 and c&3 -- all distinct mod 4.
    }
    asm volatile("s_wait_asynccnt 0x5" ::: "memory");
    __syncthreads();
    compute(cend - 2);
    asm volatile("s_wait_asynccnt 0x0" ::: "memory");
    __syncthreads();
    compute(cend - 1);

    // ---- epilogue: y[m, f] += acc * mu2[f]  (bias pre-seeded) ----
    const float m2 = mu2[f];
    const int mrow = (lane >> 4) << 3;          // C-layout: lanes 16-31 hold M+8
    #pragma unroll
    for (int r = 0; r < 8; ++r) {
        atomicAdd(out + (size_t)( 0 + mrow + r) * KDIM + f, acc0[r] * m2);
        atomicAdd(out + (size_t)(16 + mrow + r) * KDIM + f, acc1[r] * m2);
        atomicAdd(out + (size_t)(32 + mrow + r) * KDIM + f, acc2[r] * m2);
        atomicAdd(out + (size_t)(48 + mrow + r) * KDIM + f, acc3[r] * m2);
    }
}

extern "C" void kernel_launch(void* const* d_in, const int* in_sizes, int n_in,
                              void* d_out, int out_size, void* d_ws, size_t ws_size,
                              hipStream_t stream) {
    const float*  x      = (const float*)d_in[0];
    const int8_t* Wq     = (const int8_t*)d_in[1];   // 4-bit values in int8
    const float*  scales = (const float*)d_in[2];
    const float*  zeros  = (const float*)d_in[3];
    const float*  mask   = (const float*)d_in[4];
    const float*  mu1    = (const float*)d_in[5];
    const float*  mu2    = (const float*)d_in[6];
    const float*  bias   = (const float*)d_in[7];
    float*    out = (float*)d_out;
    _Float16* xh  = (_Float16*)d_ws;   // 1 MB pre-swizzled f16 activations

    // fold mu1 into x, swizzle to WMMA A-fragment layout
    prep_x_kernel<<<dim3(262144 / 256), 256, 0, stream>>>(x, mu1, xh);
    // seed output with bias
    init_out_kernel<<<dim3(524288 / 256), 256, 0, stream>>>(bias, out);
    // streaming dequant + WMMA GEMM with deep async LDS pipeline
    qlinear_wmma_kernel<<<dim3(KDIM / KPERBLOCK, NSPLIT), 128, 0, stream>>>(
        Wq, scales, zeros, mask, mu2, xh, out);
}
